// RNN_MultiRegional_4724464026158
// MI455X (gfx1250) — matrix operand order repaired and checked by reading, not compiled, not run on hardware
//
#include <hip/hip_runtime.h>
#include <hip/hip_bf16.h>

#define HID 1024
#define H3  3072
#define BB  128
#define TT  256
#define NINH 307              // int(0.3*1024)
#define EXC (HID - NINH)      // 717

typedef __attribute__((ext_vector_type(16))) __bf16 v16bf;
typedef __attribute__((ext_vector_type(8)))  __bf16 v8bf;
typedef __attribute__((ext_vector_type(8)))  float  v8f;
typedef __attribute__((ext_vector_type(4)))  int    v4i;

// Async global->LDS staging if the toolchain declares the gfx1250 builtin.
#ifndef USE_ASYNC_LDS
#if defined(__has_builtin)
#if __has_builtin(__builtin_amdgcn_global_load_async_to_lds_b128)
#define USE_ASYNC_LDS 1
#endif
#endif
#endif
#ifndef USE_ASYNC_LDS
#define USE_ASYNC_LDS 0
#endif

#if defined(__has_builtin)
#if __has_builtin(__builtin_amdgcn_s_wait_asynccnt)
#define WAIT_ASYNC() __builtin_amdgcn_s_wait_asynccnt(0)
#endif
#endif
#ifndef WAIT_ASYNC
#define WAIT_ASYNC() asm volatile("s_wait_asynccnt 0" ::: "memory")
#endif

__device__ __forceinline__ float clipw(float v) {
    v = fminf(v, 1.0f);
    return fmaxf(v, 1e-10f);
}

__device__ __forceinline__ v16bf make16(v8bf lo, v8bf hi) {
    v16bf r;
    #pragma unroll
    for (int q = 0; q < 8; ++q) { r[q] = lo[q]; r[q + 8] = hi[q]; }
    return r;
}

// ---------------------------------------------------------------------------
// Build W_rec (row i = output/N, col j = K) in bf16.  W16[i*H3 + j].
// ---------------------------------------------------------------------------
__global__ __launch_bounds__(256) void k_build_w(
    const float* __restrict__ s2t_w, const float* __restrict__ m2m_w,
    const float* __restrict__ m2s_w, const float* __restrict__ t2m_w,
    const float* __restrict__ s2s_fix, __bf16* __restrict__ W16)
{
    size_t idx = (size_t)blockIdx.x * 256 + threadIdx.x;
    if (idx >= (size_t)H3 * H3) return;
    int i = (int)(idx / H3);
    int j = (int)(idx % H3);
    float w = 0.0f;
    if (i < HID) {
        if (j < HID) {
            w = -s2s_fix[(size_t)i * HID + j];          // mask*clip term is 0; d = -1
        } else if (j >= 2 * HID) {
            int jj = j - 2 * HID;
            w = (jj < EXC) ? clipw(m2s_w[(size_t)i * HID + jj]) : 0.0f;
        }
    } else if (i < 2 * HID) {
        int ii = i - HID;
        if (j < HID) {
            float d = (j < HID / 2) ? 1.0f : -1.0f;
            w = clipw(s2t_w[(size_t)ii * HID + j]) * d;
        }
    } else {
        int ii = i - 2 * HID;
        if (j >= HID && j < 2 * HID) {
            w = clipw(t2m_w[(size_t)ii * HID + (j - HID)]);
        } else if (j >= 2 * HID) {
            int jj = j - 2 * HID;
            float d = (jj < EXC) ? 1.0f : -1.0f;
            w = clipw(m2m_w[(size_t)ii * HID + jj]) * d;
        }
    }
    W16[idx] = (__bf16)w;
}

__global__ __launch_bounds__(256) void k_init_h(const float* __restrict__ h0,
                                                __bf16* __restrict__ hbf)
{
    int e = blockIdx.x * 256 + threadIdx.x;
    if (e < BB * H3) hbf[e] = (__bf16)h0[e];
}

// ---------------------------------------------------------------------------
// One scan step:  out = relu(0.9*x0 + 0.1*(h @ W_rec^T + inp_proj_t))
// grid = (12 n-blocks of 256 cols, 4 m-blocks of 32 batch rows), 256 thr.
// Each wave: 2M x 2N register tile (4 WMMA per 4 ds + 4 vmem loads).
// A strip (32 rows x 512 K, bf16) staged in LDS per K segment (async path).
// K loop walks only the nonzero W_rec column blocks for this N region.
// ---------------------------------------------------------------------------
__global__ __launch_bounds__(256) void k_step(
    const __bf16* __restrict__ hbf_in, __bf16* __restrict__ hbf_out,
    const __bf16* __restrict__ W16, const float* __restrict__ x0,
    const float* __restrict__ inp, const float* __restrict__ inpW,
    float* __restrict__ rnn_out, int t)
{
    constexpr int LSTR = 520;                  // padded row stride (bf16)
    __shared__ __bf16 lds[32 * LSTR];          // 33280 B

    const int tid  = threadIdx.x;
    const int lane = tid & 31;
    const int wave = tid >> 5;
    const int m0    = blockIdx.y * 32;         // 32 batch rows per block
    const int nbase = blockIdx.x * 256;        // 256 output cols per block
    const int n0    = nbase + wave * 32;       // this wave's 2 N tiles
    const int region = nbase >> 10;            // 0:str 1:thal 2:m1

    int nsegs, segs[4];
    if (region == 0)      { nsegs = 4; segs[0] = 0;    segs[1] = 512;
                                       segs[2] = 2048; segs[3] = 2560; }
    else if (region == 1) { nsegs = 2; segs[0] = 0;    segs[1] = 512;
                                       segs[2] = 0;    segs[3] = 0;    }
    else                  { nsegs = 4; segs[0] = 1024; segs[1] = 1536;
                                       segs[2] = 2048; segs[3] = 2560; }

    const int  arow  = lane & 15;
    const int  hioff = (lane >= 16) ? 8 : 0;   // A-operand lane K offset
    const int  bkoff = (lane >= 16) ? 8 : 0;   // B-operand lane K offset
    const __bf16* brow0 = W16 + (size_t)(n0 + (lane & 15)) * H3;

    v8f acc[2][2] = {};

    for (int s = 0; s < nsegs; ++s) {
        const int kseg = segs[s];
        __syncthreads();
        // stage A strip: 32 rows x 512 bf16 = 2048 uint4, 8 per thread
        #pragma unroll
        for (int it = 0; it < 8; ++it) {
            int e = tid + it * 256;            // 0..2047
            int r = e >> 6;                    // 64 uint4 per row
            int c = e & 63;
            const uint4* src =
                (const uint4*)(hbf_in + (size_t)(m0 + r) * H3 + kseg) + c;
            uint4* dst = (uint4*)(lds + r * LSTR) + c;
#if USE_ASYNC_LDS
            typedef __attribute__((address_space(1))) v4i gv4i;   // __device__ v4i*
            typedef __attribute__((address_space(3))) v4i sv4i;   // __shared__ v4i*
            __builtin_amdgcn_global_load_async_to_lds_b128(
                (gv4i*)(uintptr_t)src, (sv4i*)(uint32_t)(uintptr_t)dst, 0, 0);
#else
            *dst = *src;
#endif
        }
#if USE_ASYNC_LDS
        WAIT_ASYNC();
#endif
        __syncthreads();

        #pragma unroll 2
        for (int kt = 0; kt < 16; ++kt) {
            const int k0 = kt * 32;
            // A tiles (2 M sub-tiles) from LDS
            const __bf16* ap0 = lds + arow * LSTR + k0 + hioff;
            const __bf16* ap1 = ap0 + 16 * LSTR;
            v16bf a0 = make16(*(const v8bf*)(ap0), *(const v8bf*)(ap0 + 16));
            v16bf a1 = make16(*(const v8bf*)(ap1), *(const v8bf*)(ap1 + 16));
            // B tiles (2 N sub-tiles) from global W16 (L2 resident)
            const __bf16* bp0 = brow0 + kseg + k0 + bkoff;
            const __bf16* bp1 = bp0 + (size_t)16 * H3;
            v16bf b0 = make16(*(const v8bf*)(bp0), *(const v8bf*)(bp0 + 16));
            v16bf b1 = make16(*(const v8bf*)(bp1), *(const v8bf*)(bp1 + 16));
            acc[0][0] = __builtin_amdgcn_wmma_f32_16x16x32_bf16(
                false, a0, false, b0, (short)0, acc[0][0], false, false);
            acc[0][1] = __builtin_amdgcn_wmma_f32_16x16x32_bf16(
                false, a0, false, b1, (short)0, acc[0][1], false, false);
            acc[1][0] = __builtin_amdgcn_wmma_f32_16x16x32_bf16(
                false, a1, false, b0, (short)0, acc[1][0], false, false);
            acc[1][1] = __builtin_amdgcn_wmma_f32_16x16x32_bf16(
                false, a1, false, b1, (short)0, acc[1][1], false, false);
        }
    }

    // Epilogue: fuse input projection + leaky update + relu; emit f32 + bf16.
    const int nc   = lane & 15;
    const int radd = (lane >= 16) ? 8 : 0;
    float wn[2][4];
    #pragma unroll
    for (int nt = 0; nt < 2; ++nt)
        #pragma unroll
        for (int q = 0; q < 4; ++q)
            wn[nt][q] = inpW[q * H3 + n0 + nt * 16 + nc];

    #pragma unroll
    for (int mt = 0; mt < 2; ++mt) {
        #pragma unroll
        for (int r = 0; r < 8; ++r) {
            const int bi = m0 + mt * 16 + r + radd;       // batch index
            const float4 iv = *(const float4*)(inp + ((size_t)bi * TT + t) * 4);
            #pragma unroll
            for (int nt = 0; nt < 2; ++nt) {
                const int n = n0 + nt * 16 + nc;
                const float p = iv.x * wn[nt][0] + iv.y * wn[nt][1] +
                                iv.z * wn[nt][2] + iv.w * wn[nt][3];
                float v = 0.9f * x0[(size_t)bi * H3 + n] +
                          0.1f * (acc[mt][nt][r] + p);
                v = fmaxf(v, 0.0f);
                rnn_out[((size_t)bi * TT + t) * H3 + n] = v;
                hbf_out[(size_t)bi * H3 + n] = (__bf16)v;
            }
        }
    }
}

// mean/std heads: reduce last HID cols of rnn_out per (b,t)
__global__ __launch_bounds__(256) void k_heads(
    const float* __restrict__ rnn,
    const float* __restrict__ meanW, const float* __restrict__ meanB,
    const float* __restrict__ stdW,  const float* __restrict__ stdB,
    float* __restrict__ mean_out, float* __restrict__ std_out)
{
    const int bt = blockIdx.x;
    const float* row = rnn + (size_t)bt * H3 + 2 * HID;
    __shared__ float sm[256], ss[256];
    float m = 0.0f, s = 0.0f;
    for (int nn = threadIdx.x; nn < HID; nn += 256) {
        float v = row[nn];
        m += v * meanW[2 * HID + nn];
        s += v * stdW[2 * HID + nn];
    }
    sm[threadIdx.x] = m; ss[threadIdx.x] = s;
    __syncthreads();
    for (int o = 128; o > 0; o >>= 1) {
        if (threadIdx.x < o) {
            sm[threadIdx.x] += sm[threadIdx.x + o];
            ss[threadIdx.x] += ss[threadIdx.x + o];
        }
        __syncthreads();
    }
    if (threadIdx.x == 0) {
        mean_out[bt] = sm[0] + meanB[0];
        std_out[bt]  = ss[0] + stdB[0];
    }
}

// x_out = broadcast of x0 over T (float4 streaming stores)
__global__ __launch_bounds__(256) void k_xout(const float* __restrict__ x0,
                                              float* __restrict__ xout)
{
    size_t e4 = (size_t)blockIdx.x * 256 + threadIdx.x;
    const size_t total4 = (size_t)BB * TT * H3 / 4;
    if (e4 >= total4) return;
    size_t e = e4 * 4;
    size_t n = e % H3;
    size_t b = (e / H3) / TT;
    *(float4*)(xout + e) = *(const float4*)(x0 + b * H3 + n);
}

__global__ __launch_bounds__(256) void k_lasts(const float* __restrict__ x0,
                                               const float* __restrict__ rnn,
                                               float* __restrict__ hn_last,
                                               float* __restrict__ x_last)
{
    int e = blockIdx.x * 256 + threadIdx.x;
    if (e >= BB * H3) return;
    int b = e / H3, n = e % H3;
    hn_last[e] = rnn[((size_t)b * TT + (TT - 1)) * H3 + n];
    x_last[e]  = x0[e];
}

extern "C" void kernel_launch(void* const* d_in, const int* in_sizes, int n_in,
                              void* d_out, int out_size, void* d_ws, size_t ws_size,
                              hipStream_t stream)
{
    (void)in_sizes; (void)n_in; (void)out_size; (void)ws_size;
    const float* inp     = (const float*)d_in[0];
    const float* hn      = (const float*)d_in[1];
    const float* x0      = (const float*)d_in[2];
    // d_in[3] = str2str_w (multiplied by zero mask -> unused)
    const float* s2t_w   = (const float*)d_in[4];
    const float* m2m_w   = (const float*)d_in[5];
    const float* m2s_w   = (const float*)d_in[6];
    const float* t2m_w   = (const float*)d_in[7];
    const float* s2s_fix = (const float*)d_in[8];
    const float* inpW    = (const float*)d_in[9];
    const float* meanW   = (const float*)d_in[10];
    const float* meanB   = (const float*)d_in[11];
    const float* stdW    = (const float*)d_in[12];
    const float* stdB    = (const float*)d_in[13];

    float* out = (float*)d_out;
    // output tuple offsets (floats): mean, std, rnn_out, hn_last, x_last, x_out
    const size_t o_std   = (size_t)BB * TT;
    const size_t o_rnn   = o_std * 2;
    const size_t o_hn    = o_rnn + (size_t)BB * TT * H3;
    const size_t o_xl    = o_hn + (size_t)BB * H3;
    const size_t o_xout  = o_xl + (size_t)BB * H3;
    float* mean_out = out;
    float* std_out  = out + o_std;
    float* rnn_out  = out + o_rnn;
    float* hn_last  = out + o_hn;
    float* x_last   = out + o_xl;
    float* x_out    = out + o_xout;

    char* ws = (char*)d_ws;
    __bf16* W16  = (__bf16*)ws;                                   // 18.87 MB
    __bf16* hbf0 = (__bf16*)(ws + (size_t)H3 * H3 * 2);           // 768 KB
    __bf16* hbf1 = hbf0 + (size_t)BB * H3;                        // 768 KB

    k_build_w<<<(H3 * H3 + 255) / 256, 256, 0, stream>>>(
        s2t_w, m2m_w, m2s_w, t2m_w, s2s_fix, W16);
    k_init_h<<<(BB * H3 + 255) / 256, 256, 0, stream>>>(hn, hbf0);

    dim3 sg(12, 4);
    for (int t = 0; t < TT; ++t) {
        const __bf16* hin = (t & 1) ? hbf1 : hbf0;
        __bf16*      hout = (t & 1) ? hbf0 : hbf1;
        k_step<<<sg, 256, 0, stream>>>(hin, hout, W16, x0, inp, inpW, rnn_out, t);
    }

    k_heads<<<BB * TT, 256, 0, stream>>>(rnn_out, meanW, meanB, stdW, stdB,
                                         mean_out, std_out);
    k_xout<<<((size_t)BB * TT * H3 / 4 + 255) / 256, 256, 0, stream>>>(x0, x_out);
    k_lasts<<<(BB * H3 + 255) / 256, 256, 0, stream>>>(x0, rnn_out, hn_last, x_last);
}
